// Qwen3SeqParallelMLP_27951647162501
// MI455X (gfx1250) — compile-verified, matrix operand order checked by reading, not run on hardware
//
#include <hip/hip_runtime.h>
#include <stdint.h>

// ---------------- types ----------------
typedef __bf16 bf16x16 __attribute__((ext_vector_type(16)));
typedef float  v8f     __attribute__((ext_vector_type(8)));

// ---------------- problem dims (reference) ----------------
#define HIDDEN 2048
#define INTER  8192
#define MROWS  8192   // BATCH * LOCAL_SEQ = 2 * 4096

// ---------------- tiling ----------------
#define BM  128
#define BN  64
#define BK1 32        // GEMM1 K-tile
#define BK2 64        // GEMM2 K-tile
#define NTHREADS 256  // 8 waves (wave32)

// fp32 pair -> packed bf16 via native V_CVT_PK_BF16_F32 (single VALU op).
// The clang builtin is not declared on this toolchain, so emit directly.
__device__ __forceinline__ uint32_t pack2(float a, float b) {
    uint32_t r;
    asm("v_cvt_pk_bf16_f32 %0, %1, %2" : "=v"(r) : "v"(a), "v"(b));
    return r;
}
__device__ __forceinline__ uint16_t f2bf(float a) {
    return (uint16_t)pack2(a, a);
}

__device__ __forceinline__ float fast_rcp(float x) {
#if __has_builtin(__builtin_amdgcn_rcpf)
    return __builtin_amdgcn_rcpf(x);   // v_rcp_f32
#else
    return 1.0f / x;
#endif
}

// Load one 16x32 bf16 fragment from an LDS tile with row stride `stride`.
// Lane mapping per CDNA5 ISA (16-bit A/B 16x32): lanes 0-15 -> K {0..7,16..23},
// lanes 16-31 -> K {8..15,24..31}; two ds_load_b128 per lane.
struct Bits256 { uint4 lo, hi; };
__device__ __forceinline__ bf16x16 frag_lds(const uint16_t* base, int row, int stride, int kb) {
    Bits256 t;
    t.lo = *(const uint4*)(base + row * stride + kb);
    t.hi = *(const uint4*)(base + row * stride + kb + 16);
    return __builtin_bit_cast(bf16x16, t);
}

// CDNA5 async global->LDS copy (ASYNCcnt-tracked), per CDNA5 ISA.
// VDST VGPR = LDS byte address (generic shared addr low 32 bits == LDS offset).
__device__ __forceinline__ void async_g2l_b128(uint32_t lds_addr, const void* gaddr) {
    asm volatile("global_load_async_to_lds_b128 %0, %1, off"
                 :: "v"(lds_addr), "v"(gaddr) : "memory");
}
__device__ __forceinline__ void wait_asynccnt0() {
    asm volatile("s_wait_asynccnt 0" ::: "memory");
}

// =====================================================================
// GEMM1: act[m, n] = silu(h @ Wg^T) * (h @ Wu^T), act stored as bf16 bits
//   h   : [MROWS, HIDDEN] fp32
//   wgu : [2*INTER, HIDDEN] fp32 (rows 0..INTER-1 = gate, INTER.. = up)
// Double-buffered LDS, one barrier per K-tile, fp32->bf16 convert on the fly.
// =====================================================================
__global__ __launch_bounds__(NTHREADS) void gemm_gate_up_silu(
    const float* __restrict__ h, const float* __restrict__ wgu,
    uint16_t* __restrict__ act)
{
    __shared__ __align__(16) uint16_t lA[2][BM * BK1];      // 2 x 8 KB
    __shared__ __align__(16) uint16_t lB[2][2 * BN * BK1];  // 2 x 8 KB (gate|up)

    const int tid  = threadIdx.x;
    const int n0   = blockIdx.x * BN;
    const int m0   = blockIdx.y * BM;

    const int lane  = tid & 31;
    const int wave  = tid >> 5;
    const int waveM = wave & 3;         // 4 row-strips of 32
    const int waveN = wave >> 2;        // 2 col-strips of 32
    const int half  = lane >> 4;
    const int lidx  = lane & 15;
    const int kb    = half * 8;

    // loader assignment: 2 threads per row, 16 floats each
    const int lr = tid >> 1;            // 0..127
    const int lk = (tid & 1) * 16;
    const int wrow = (lr < BN) ? (n0 + lr) : (INTER + n0 + (lr - BN));
    const float* pA = h   + (size_t)(m0 + lr) * HIDDEN + lk;
    const float* pB = wgu + (size_t)wrow      * HIDDEN + lk;

    float4 a0, a1, a2, a3, b0, b1, b2, b3;
    auto load_regs = [&](int k0) {
        const float4* ga = (const float4*)(pA + k0);
        a0 = ga[0]; a1 = ga[1]; a2 = ga[2]; a3 = ga[3];
        const float4* gb = (const float4*)(pB + k0);
        b0 = gb[0]; b1 = gb[1]; b2 = gb[2]; b3 = gb[3];
    };
    auto store_lds = [&](int buf) {
        uint32_t* pa = (uint32_t*)&lA[buf][lr * BK1 + lk];
        pa[0] = pack2(a0.x, a0.y); pa[1] = pack2(a0.z, a0.w);
        pa[2] = pack2(a1.x, a1.y); pa[3] = pack2(a1.z, a1.w);
        pa[4] = pack2(a2.x, a2.y); pa[5] = pack2(a2.z, a2.w);
        pa[6] = pack2(a3.x, a3.y); pa[7] = pack2(a3.z, a3.w);
        uint32_t* pb = (uint32_t*)&lB[buf][lr * BK1 + lk];
        pb[0] = pack2(b0.x, b0.y); pb[1] = pack2(b0.z, b0.w);
        pb[2] = pack2(b1.x, b1.y); pb[3] = pack2(b1.z, b1.w);
        pb[4] = pack2(b2.x, b2.y); pb[5] = pack2(b2.z, b2.w);
        pb[6] = pack2(b3.x, b3.y); pb[7] = pack2(b3.z, b3.w);
    };

    v8f accG[2][2], accU[2][2];
#pragma unroll
    for (int i = 0; i < 2; ++i)
#pragma unroll
        for (int j = 0; j < 2; ++j) { accG[i][j] = (v8f)0.0f; accU[i][j] = (v8f)0.0f; }

    load_regs(0);
    store_lds(0);

    const int nIter = HIDDEN / BK1;
    for (int i = 0; i < nIter; ++i) {
        const int cur = i & 1;
        if (i + 1 < nIter) load_regs((i + 1) * BK1);
        if (i + 2 < nIter) __builtin_prefetch(pB + (i + 2) * BK1, 0, 3);
        __syncthreads();   // publishes buffer `cur`; all reads of cur^1 done

        bf16x16 af[2], bg[2], bu[2];
#pragma unroll
        for (int tm = 0; tm < 2; ++tm)
            af[tm] = frag_lds(&lA[cur][0], waveM * 32 + tm * 16 + lidx, BK1, kb);
#pragma unroll
        for (int tn = 0; tn < 2; ++tn) {
            bg[tn] = frag_lds(&lB[cur][0], waveN * 32 + tn * 16 + lidx, BK1, kb);
            bu[tn] = frag_lds(&lB[cur][0], BN + waveN * 32 + tn * 16 + lidx, BK1, kb);
        }
#pragma unroll
        for (int tm = 0; tm < 2; ++tm)
#pragma unroll
            for (int tn = 0; tn < 2; ++tn) {
                accG[tm][tn] = __builtin_amdgcn_wmma_f32_16x16x32_bf16(
                    false, af[tm], false, bg[tn], (short)0, accG[tm][tn], false, false);
                accU[tm][tn] = __builtin_amdgcn_wmma_f32_16x16x32_bf16(
                    false, af[tm], false, bu[tn], (short)0, accU[tm][tn], false, false);
            }

        if (i + 1 < nIter) store_lds(cur ^ 1);
    }

    // ---- fused SiLU(gate)*up epilogue, store bf16 act ----
#pragma unroll
    for (int tm = 0; tm < 2; ++tm)
#pragma unroll
        for (int tn = 0; tn < 2; ++tn) {
            v8f g = accG[tm][tn], u = accU[tm][tn];
            const int n  = n0 + waveN * 32 + tn * 16 + lidx;
            const int mb = m0 + waveM * 32 + tm * 16 + half * 8;
#pragma unroll
            for (int r = 0; r < 8; ++r) {
                float gv = g[r];
                float s  = gv * fast_rcp(1.0f + __expf(-gv));   // silu via v_rcp_f32
                float a  = s * u[r];
                act[(size_t)(mb + r) * INTER + n] = f2bf(a);
            }
        }
}

// =====================================================================
// GEMM2: out[m, n] = act @ Wd^T
//   act : [MROWS, INTER] bf16 bits  -> A tile via async global->LDS copy
//   wd  : [HIDDEN, INTER] fp32      -> B tile converted in VGPRs
//   out : [MROWS, HIDDEN] fp32
// BK=64, double-buffered, one barrier per K-tile.
// =====================================================================
__global__ __launch_bounds__(NTHREADS) void gemm_down(
    const uint16_t* __restrict__ act, const float* __restrict__ wd,
    float* __restrict__ out)
{
    __shared__ __align__(16) uint16_t lA[2][BM * BK2];  // 2 x 16 KB
    __shared__ __align__(16) uint16_t lB[2][BN * BK2];  // 2 x  8 KB

    const int tid = threadIdx.x;
    const int n0  = blockIdx.x * BN;
    const int m0  = blockIdx.y * BM;

    const int lane  = tid & 31;
    const int wave  = tid >> 5;
    const int waveM = wave & 3;
    const int waveN = wave >> 2;
    const int half  = lane >> 4;
    const int lidx  = lane & 15;
    const int kb    = half * 8;

    // A loader: 2 threads/row, 32 bf16 (64 B) each -> 4 async b128 copies
    const int ar  = tid >> 1;           // 0..127
    const int ako = (tid & 1) * 32;
    // B loader: 4 threads/row, 16 fp32 each
    const int br  = tid >> 2;           // 0..63
    const int bko = (tid & 3) * 16;

    const uint16_t* pA = act + (size_t)(m0 + ar) * INTER + ako;
    const float*    pB = wd  + (size_t)(n0 + br) * INTER + bko;

    auto async_a = [&](int k0, int buf) {
        uint32_t l = (uint32_t)(uintptr_t)&lA[buf][ar * BK2 + ako];
        const uint16_t* g = pA + k0;
        async_g2l_b128(l,      g);
        async_g2l_b128(l + 16, g + 8);
        async_g2l_b128(l + 32, g + 16);
        async_g2l_b128(l + 48, g + 24);
    };

    float4 b0, b1, b2, b3;
    auto load_b_regs = [&](int k0) {
        const float4* gb = (const float4*)(pB + k0);
        b0 = gb[0]; b1 = gb[1]; b2 = gb[2]; b3 = gb[3];
    };
    auto store_b_lds = [&](int buf) {
        uint32_t* pb = (uint32_t*)&lB[buf][br * BK2 + bko];
        pb[0] = pack2(b0.x, b0.y); pb[1] = pack2(b0.z, b0.w);
        pb[2] = pack2(b1.x, b1.y); pb[3] = pack2(b1.z, b1.w);
        pb[4] = pack2(b2.x, b2.y); pb[5] = pack2(b2.z, b2.w);
        pb[6] = pack2(b3.x, b3.y); pb[7] = pack2(b3.z, b3.w);
    };

    v8f acc[2][2];
#pragma unroll
    for (int i = 0; i < 2; ++i)
#pragma unroll
        for (int j = 0; j < 2; ++j) acc[i][j] = (v8f)0.0f;

    // prologue: tile 0 into buffer 0
    async_a(0, 0);
    load_b_regs(0);
    store_b_lds(0);
    wait_asynccnt0();

    const int nIter = INTER / BK2;      // 128
    for (int i = 0; i < nIter; ++i) {
        const int cur = i & 1;
        __syncthreads();   // publishes buffer `cur`; all reads of cur^1 done

        if (i + 1 < nIter) {
            async_a((i + 1) * BK2, cur ^ 1);   // issued AFTER barrier -> safe vs readers
            load_b_regs((i + 1) * BK2);
        }
        if (i + 2 < nIter) __builtin_prefetch(pB + (i + 2) * BK2, 0, 3);

#pragma unroll
        for (int ks = 0; ks < BK2; ks += 32) {
            bf16x16 af[2], bf[2];
#pragma unroll
            for (int tm = 0; tm < 2; ++tm)
                af[tm] = frag_lds(&lA[cur][0], waveM * 32 + tm * 16 + lidx, BK2, ks + kb);
#pragma unroll
            for (int tn = 0; tn < 2; ++tn)
                bf[tn] = frag_lds(&lB[cur][0], waveN * 32 + tn * 16 + lidx, BK2, ks + kb);
#pragma unroll
            for (int tm = 0; tm < 2; ++tm)
#pragma unroll
                for (int tn = 0; tn < 2; ++tn)
                    acc[tm][tn] = __builtin_amdgcn_wmma_f32_16x16x32_bf16(
                        false, af[tm], false, bf[tn], (short)0, acc[tm][tn], false, false);
        }

        if (i + 1 < nIter) {
            store_b_lds(cur ^ 1);
            wait_asynccnt0();   // async writes to cur^1 complete before next barrier
        }
    }

    // ---- fp32 epilogue ----
#pragma unroll
    for (int tm = 0; tm < 2; ++tm)
#pragma unroll
        for (int tn = 0; tn < 2; ++tn) {
            v8f c = acc[tm][tn];
            const int n  = n0 + waveN * 32 + tn * 16 + lidx;
            const int mb = m0 + waveM * 32 + tm * 16 + half * 8;
#pragma unroll
            for (int r = 0; r < 8; ++r)
                out[(size_t)(mb + r) * HIDDEN + n] = c[r];
        }
}

// =====================================================================
extern "C" void kernel_launch(void* const* d_in, const int* in_sizes, int n_in,
                              void* d_out, int out_size, void* d_ws, size_t ws_size,
                              hipStream_t stream) {
    (void)in_sizes; (void)n_in; (void)out_size; (void)ws_size;
    const float* h   = (const float*)d_in[0];   // [2,4096,2048] fp32
    const float* wgu = (const float*)d_in[1];   // [16384,2048]  fp32
    const float* wd  = (const float*)d_in[2];   // [2048,8192]   fp32
    float*    out = (float*)d_out;              // [2,4096,2048] fp32
    uint16_t* act = (uint16_t*)d_ws;            // [8192,8192] bf16 bits (128 MiB)

    dim3 g1(INTER / BN, MROWS / BM);            // (128, 64)
    gemm_gate_up_silu<<<g1, NTHREADS, 0, stream>>>(h, wgu, act);

    dim3 g2(HIDDEN / BN, MROWS / BM);           // (32, 64)
    gemm_down<<<g2, NTHREADS, 0, stream>>>(act, wd, out);
}
